// EDNCustomRNN_7146825580591
// MI455X (gfx1250) — matrix operand order repaired and checked by reading, not compiled, not run on hardware
//
#include <hip/hip_runtime.h>

// ---------------------------------------------------------------------------
// Bidirectional LSTM + projection for MI455X (gfx1250, wave32, WMMA bf16).
// B=128, T=256, I=512, H=256, O=256.
//
// Pipeline (all on `stream`, serialized by stream order):
//   1) cvt: fp32 -> bf16 for x, w_ih_{f,b}, w_hh_{f,b}, w_lin   (into ws)
//   2) wmma_gemm (ping-pong pipelined): xg_d = x_bf @ w_ih_d^T + (b_ih+b_hh)
//   3) lstm_scan: 16 workgroups (2 dirs x 8 batch tiles of 16 rows),
//      h/c state in LDS, per-step gates = xg_t + h[16x256] @ w_hh^T via WMMA,
//      w_hh B-tiles register-pipelined, next-step xg prefetched.
//   4) wmma_gemm: out = rec_bf @ w_lin^T + b_lin
// ---------------------------------------------------------------------------

typedef __bf16 bf16;
typedef __attribute__((ext_vector_type(8)))  __bf16 bf16x8;
typedef __attribute__((ext_vector_type(16))) __bf16 v16bf;
typedef __attribute__((ext_vector_type(8)))  float  v8f;

#define CB 128
#define CT 256
#define CI 512
#define CH 256

static __device__ __forceinline__ v16bf cat16(bf16x8 lo, bf16x8 hi) {
  return __builtin_shufflevector(lo, hi, 0,1,2,3,4,5,6,7,8,9,10,11,12,13,14,15);
}
static __device__ __forceinline__ float sigf(float x) {
  return 1.0f / (1.0f + __expf(-x));
}

// --------------------------- fp32 -> bf16 convert ---------------------------
__global__ void cvt_f32_bf16(const float* __restrict__ src, bf16* __restrict__ dst, int n) {
  int i = blockIdx.x * blockDim.x + threadIdx.x;
  if (i < n) dst[i] = (bf16)src[i];
}

// ------------------------------ WMMA GEMM ----------------------------------
// out[M,N] (fp32) = A[M,K](bf16) * Bw[N,K](bf16)^T + bias0[N] (+ bias1[N])
// Block = 128 threads = 4 waves; wave computes a 16(M) x 64(N) tile.
// k-loop unrolled x2 with ping-pong register buffers: loads for stage i+2
// land directly in the buffer consumed later -> no loop-carried reg copies,
// WMMAs of stage i overlap the in-flight loads of stage i+1.
// Requires K % 64 == 0 (true for all call sites: K=512).
// grid.x = M/16, grid.y = N/256.
__global__ __launch_bounds__(128)
void wmma_gemm(const bf16* __restrict__ A, const bf16* __restrict__ Bw,
               const float* __restrict__ bias0, const float* __restrict__ bias1,
               float* __restrict__ out, int M, int N, int K) {
  const int lane = threadIdx.x & 31;
  const int wave = threadIdx.x >> 5;
  const int hi   = lane >> 4;       // 0: lanes 0-15, 1: lanes 16-31
  const int ln   = lane & 15;
  const int m0   = blockIdx.x * 16;
  const int n0   = (blockIdx.y * 4 + wave) * 64;
  if (m0 >= M || n0 >= N) return;   // wave-uniform: EXEC stays all ones

  v8f acc[4];
#pragma unroll
  for (int nt = 0; nt < 4; ++nt) {
    const int col = n0 + nt * 16 + ln;
    float bv = bias0 ? bias0[col] : 0.0f;
    if (bias1) bv += bias1[col];
#pragma unroll
    for (int r = 0; r < 8; ++r) acc[nt][r] = bv;
  }

  // A tile: lanes 0-15 -> row M=ln, K=0..15; lanes 16-31 -> K=16..31
  const bf16* __restrict__ ap = A + (size_t)(m0 + ln) * K + (hi ? 16 : 0);
  // B tile: lane -> column N; lanes 0-15 K=0..15, lanes 16-31 K=16..31
  const bf16* __restrict__ bp[4];
#pragma unroll
  for (int nt = 0; nt < 4; ++nt)
    bp[nt] = Bw + (size_t)(n0 + nt * 16 + ln) * K + (hi ? 16 : 0);

  auto ldA = [&](int k) { return cat16(*(const bf16x8*)(ap + k),
                                       *(const bf16x8*)(ap + k + 8)); };
  auto ldB = [&](int nt, int k) { return cat16(*(const bf16x8*)(bp[nt] + k),
                                               *(const bf16x8*)(bp[nt] + k + 8)); };

  // prologue: stage 0 buffers
  v16bf a0 = ldA(0);
  v16bf b0[4];
#pragma unroll
  for (int nt = 0; nt < 4; ++nt) b0[nt] = ldB(nt, 0);

  for (int k0 = 0; k0 < K; k0 += 64) {
    // prefetch stage 1 (k0+32)
    v16bf a1 = ldA(k0 + 32);
    v16bf b1[4];
#pragma unroll
    for (int nt = 0; nt < 4; ++nt) b1[nt] = ldB(nt, k0 + 32);
    // compute stage 0
#pragma unroll
    for (int nt = 0; nt < 4; ++nt)
      acc[nt] = __builtin_amdgcn_wmma_f32_16x16x32_bf16(
          false, a0, false, b0[nt], (short)0, acc[nt], false, false);
    // prefetch stage 2 (k0+64) straight into the stage-0 buffers
    const int kn = (k0 + 64 < K) ? (k0 + 64) : 0;   // harmless re-read on last iter
    a0 = ldA(kn);
#pragma unroll
    for (int nt = 0; nt < 4; ++nt) b0[nt] = ldB(nt, kn);
    // compute stage 1
#pragma unroll
    for (int nt = 0; nt < 4; ++nt)
      acc[nt] = __builtin_amdgcn_wmma_f32_16x16x32_bf16(
          false, a1, false, b1[nt], (short)0, acc[nt], false, false);
  }

  const int rbase = hi * 8;          // C/D: lanes 0-15 rows 0-7, lanes 16-31 rows 8-15
#pragma unroll
  for (int nt = 0; nt < 4; ++nt) {
    const int col = n0 + nt * 16 + ln;
#pragma unroll
    for (int r = 0; r < 8; ++r)
      out[(size_t)(m0 + rbase + r) * N + col] = acc[nt][r];
  }
}

// ------------------------------ LSTM scan ----------------------------------
// grid.x = 16: dir = bx>>3, batch tile = (bx&7)*16. Block = 256 thr = 8 waves.
// Wave w owns gate columns [w*128, w*128+128). h (bf16) and c (f32) in LDS.
__global__ __launch_bounds__(256)
void lstm_scan(const float* __restrict__ xg_f, const float* __restrict__ xg_b,
               const bf16* __restrict__ whh_f, const bf16* __restrict__ whh_b,
               bf16* __restrict__ rec) {
  const int dir = blockIdx.x >> 3;
  const int b0  = (blockIdx.x & 7) * 16;
  const float* __restrict__ xg  = dir ? xg_b  : xg_f;
  const bf16*  __restrict__ whh = dir ? whh_b : whh_f;

  __shared__ __align__(16) bf16  h_lds[16][CH];      //  8 KB
  __shared__ __align__(16) bf16  g_lds[16][4 * CH];  // 32 KB (activated gates)
  __shared__ __align__(16) float c_lds[16][CH];      // 16 KB

  for (int idx = threadIdx.x; idx < 16 * CH; idx += 256) {
    (&h_lds[0][0])[idx] = (bf16)0.0f;
    (&c_lds[0][0])[idx] = 0.0f;
  }
  __syncthreads();

  const int lane = threadIdx.x & 31;
  const int wave = threadIdx.x >> 5;
  const int hi   = lane >> 4;
  const int ln   = lane & 15;
  const int n0w  = wave * 128;

  // loop-invariant pointers
  const bf16* __restrict__ bpw[8];
#pragma unroll
  for (int nt = 0; nt < 8; ++nt)
    bpw[nt] = whh + (size_t)(n0w + nt * 16 + ln) * CH + (hi ? 16 : 0);
  const bf16* __restrict__ hrow = &h_lds[ln][hi ? 16 : 0];
  // xg row bases for this lane's 8 C rows (advance by t each step)
  const float* __restrict__ xrow[8];
#pragma unroll
  for (int r = 0; r < 8; ++r)
    xrow[r] = xg + (size_t)(b0 + hi * 8 + r) * CT * (4 * CH);

  auto ldA = [&](int k) { return cat16(*(const bf16x8*)(hrow + k),
                                       *(const bf16x8*)(hrow + k + 8)); };
  auto ldB = [&](int nt, int k) { return cat16(*(const bf16x8*)(bpw[nt] + k),
                                               *(const bf16x8*)(bpw[nt] + k + 8)); };

  for (int tt = 0; tt < CT; ++tt) {
    const int t = dir ? (CT - 1 - tt) : tt;

    // prefetch next timestep's xg slice (16 rows x 4KB) while we compute
    if (tt + 1 < CT) {
      const int tn = dir ? (CT - 2 - tt) : (tt + 1);
      const int pr = threadIdx.x >> 4;        // row 0..15
      const int ps = threadIdx.x & 15;        // 64-float segment 0..15
      const float* pp = xg + ((size_t)(b0 + pr) * CT + tn) * (4 * CH) + ps * 64;
      __builtin_prefetch(pp, 0, 1);
      __builtin_prefetch(pp + 32, 0, 1);
    }

    // ---- gates = xg[:, t, :] + h @ whh^T  (8 WMMA tiles per wave) ----
    v8f acc[8];
#pragma unroll
    for (int nt = 0; nt < 8; ++nt) {
      const int col = n0w + nt * 16 + ln;
#pragma unroll
      for (int r = 0; r < 8; ++r)
        acc[nt][r] = xrow[r][(size_t)t * (4 * CH) + col];
    }

    // B-tile register pipeline, one tile deep (A comes from LDS: cheap).
    // nt-loop fully unrolled -> temporaries are SSA, no register copies.
    v16bf a_cur = ldA(0);
    v16bf b_cur = ldB(0, 0);
#pragma unroll
    for (int k0 = 0; k0 < CH; k0 += 32) {
#pragma unroll
      for (int nt = 0; nt < 8; ++nt) {
        const int nt_n = (nt + 1) & 7;
        const int k_n  = (nt == 7) ? ((k0 + 32) & (CH - 1)) : k0;  // wraps; unused tail
        const v16bf b_nxt = ldB(nt_n, k_n);
        acc[nt] = __builtin_amdgcn_wmma_f32_16x16x32_bf16(
            false, a_cur, false, b_cur, (short)0, acc[nt], false, false);
        b_cur = b_nxt;
      }
      if (k0 + 32 < CH) a_cur = ldA(k0 + 32);
    }

    // ---- per-gate activation, publish to LDS ----
#pragma unroll
    for (int nt = 0; nt < 8; ++nt) {
      const int col = n0w + nt * 16 + ln;
      const bool is_g = ((col >> 8) == 2);    // gate order: i, f, g, o
#pragma unroll
      for (int r = 0; r < 8; ++r) {
        float v = acc[nt][r];
        v = is_g ? tanhf(v) : sigf(v);
        g_lds[hi * 8 + r][col] = (bf16)v;
      }
    }
    __syncthreads();

    // ---- cell/hidden update: thread j handles hidden unit j, all 16 rows ----
    {
      const int j = threadIdx.x;
      bf16* __restrict__ rbase = rec + (size_t)t * (2 * CH) + dir * CH + j;
#pragma unroll 4
      for (int r = 0; r < 16; ++r) {
        const float iv = (float)g_lds[r][j];
        const float fv = (float)g_lds[r][CH + j];
        const float gv = (float)g_lds[r][2 * CH + j];
        const float ov = (float)g_lds[r][3 * CH + j];
        const float c  = fv * c_lds[r][j] + iv * gv;
        c_lds[r][j] = c;
        const float h = ov * tanhf(c);
        h_lds[r][j] = (bf16)h;
        rbase[(size_t)(b0 + r) * CT * (2 * CH)] = (bf16)h;
      }
    }
    __syncthreads();
  }
}

// ------------------------------ launcher -----------------------------------
extern "C" void kernel_launch(void* const* d_in, const int* in_sizes, int n_in,
                              void* d_out, int out_size, void* d_ws, size_t ws_size,
                              hipStream_t stream) {
  const float* x    = (const float*)d_in[0];
  const float* wihf = (const float*)d_in[1];
  const float* whhf = (const float*)d_in[2];
  const float* bihf = (const float*)d_in[3];
  const float* bhhf = (const float*)d_in[4];
  const float* wihb = (const float*)d_in[5];
  const float* whhb = (const float*)d_in[6];
  const float* bihb = (const float*)d_in[7];
  const float* bhhb = (const float*)d_in[8];
  const float* wlin = (const float*)d_in[9];
  const float* blin = (const float*)d_in[10];
  float* out = (float*)d_out;

  // workspace layout (256B-aligned regions)
  char* ws = (char*)d_ws;
  size_t off = 0;
  auto take = [&](size_t bytes) {
    void* p = ws + off;
    off += (bytes + 255) & ~(size_t)255;
    return p;
  };
  bf16*  x_bf    = (bf16*)take((size_t)CB * CT * CI * 2);        //  32 MB
  bf16*  wihf_bf = (bf16*)take((size_t)4 * CH * CI * 2);
  bf16*  wihb_bf = (bf16*)take((size_t)4 * CH * CI * 2);
  bf16*  whhf_bf = (bf16*)take((size_t)4 * CH * CH * 2);
  bf16*  whhb_bf = (bf16*)take((size_t)4 * CH * CH * 2);
  bf16*  wlin_bf = (bf16*)take((size_t)CH * 2 * CH * 2);
  float* xg_f    = (float*)take((size_t)CB * CT * 4 * CH * 4);   // 128 MB
  float* xg_b    = (float*)take((size_t)CB * CT * 4 * CH * 4);   // 128 MB
  bf16*  rec_bf  = (bf16*)take((size_t)CB * CT * 2 * CH * 2);    //  32 MB

  auto cvt = [&](const float* s, bf16* d, int n) {
    cvt_f32_bf16<<<(n + 255) / 256, 256, 0, stream>>>(s, d, n);
  };
  cvt(x,    x_bf,    CB * CT * CI);
  cvt(wihf, wihf_bf, 4 * CH * CI);
  cvt(wihb, wihb_bf, 4 * CH * CI);
  cvt(whhf, whhf_bf, 4 * CH * CH);
  cvt(whhb, whhb_bf, 4 * CH * CH);
  cvt(wlin, wlin_bf, CH * 2 * CH);

  const int M = CB * CT;  // 32768
  // input-gate GEMMs: [32768 x 512] @ [512 x 1024]
  wmma_gemm<<<dim3(M / 16, (4 * CH) / 256), 128, 0, stream>>>(
      x_bf, wihf_bf, bihf, bhhf, xg_f, M, 4 * CH, CI);
  wmma_gemm<<<dim3(M / 16, (4 * CH) / 256), 128, 0, stream>>>(
      x_bf, wihb_bf, bihb, bhhb, xg_b, M, 4 * CH, CI);

  // recurrent scan: 2 dirs x 8 batch tiles
  lstm_scan<<<16, 256, 0, stream>>>(xg_f, xg_b, whhf_bf, whhb_bf, rec_bf);

  // projection: [32768 x 512] @ [512 x 256]
  wmma_gemm<<<dim3(M / 16, CH / 256), 128, 0, stream>>>(
      rec_bf, wlin_bf, blin, nullptr, out, M, CH, 2 * CH);
}
